// EfmLSTM_90357521973269
// MI455X (gfx1250) — matrix operand order repaired and checked by reading, not compiled, not run on hardware
//
#include <hip/hip_runtime.h>
#include <hip/hip_bf16.h>

// ---------------------------------------------------------------------------
// EfmLSTM on MI455X (gfx1250, wave32, WMMA + TDM)
//   B=256, T=1024, D=256, U=128, SIG=20, 3U=384
//
// Kernel 0 (pack):  pre-convert all weight matrices into packed bf16 WMMA
//                   B-fragments (ISA 32x16 lane layout) in workspace.
// Kernel 1 (proj):  x_proj = inputs @ Win  (K=256, bf16x3 split WMMA),
//                   f_seq = sigmoid(sigs @ Wf + b_f). A-fragments staged in
//                   LDS once per block; W-fragments are raw b128 loads.
//                   Results stored TIME-MAJOR: xp[t][b][384], fs[t][b][128].
// Kernel 2 (scan):  16 blocks x 24 waves; W_rec fragments resident in VGPRs
//                   for all 1024 steps; per-step x/f tiles DMA'd into an LDS
//                   double buffer by the Tensor Data Mover (tensor_load_to_lds
//                   + s_wait_tensorcnt), overlapping HBM with WMMA/exp work.
// ---------------------------------------------------------------------------

#define BB   256
#define TT   1024
#define DD   256
#define UU   128
#define NS   20
#define N3U  384

typedef __attribute__((ext_vector_type(16))) __bf16 v16bf;
typedef __attribute__((ext_vector_type(8)))  float  v8f;
typedef __attribute__((ext_vector_type(4)))  unsigned int v4u;
typedef __attribute__((ext_vector_type(8)))  int    v8i_t;
typedef __attribute__((ext_vector_type(4)))  int    v4i_t;

union AFrag { unsigned int u[8]; v16bf v; };
union DFrag { float f[8]; v8f v; };

// packed-fragment table sizes (u32 words)
#define WPIN_WORDS  (24 * 8 * 32 * 8)   // w_in  hi (and lo) frags
#define WREC_WORDS  (24 * 4 * 32 * 8)   // w_rec hi frags
#define WF_WORDS    (8 * 32 * 8)        // w_f   hi (and lo) frags

__device__ __forceinline__ unsigned short bf16_rn(float f) {
  unsigned int u = __float_as_uint(f);
  unsigned int r = u + 0x7FFFu + ((u >> 16) & 1u);
  return (unsigned short)(r >> 16);
}
__device__ __forceinline__ float bf16_f32(unsigned short h) {
  return __uint_as_float(((unsigned int)h) << 16);
}
__device__ __forceinline__ unsigned int pack2(unsigned short a, unsigned short b) {
  return (unsigned int)a | ((unsigned int)b << 16);
}
__device__ __forceinline__ unsigned int pack_hi(float f0, float f1) {
  return pack2(bf16_rn(f0), bf16_rn(f1));
}
__device__ __forceinline__ unsigned int pack_lo(float f0, float f1) {
  unsigned short h0 = bf16_rn(f0), h1 = bf16_rn(f1);
  return pack2(bf16_rn(f0 - bf16_f32(h0)), bf16_rn(f1 - bf16_f32(h1)));
}
// ISA A-matrix 16x32 (16-bit): k of the bf16 pair held in VGPR v, lane-half h
__device__ __forceinline__ int a_kpair(int v, int half) {
  return (v < 4) ? (2 * v + 8 * half) : (16 + 2 * (v - 4) + 8 * half);
}
__device__ __forceinline__ float fast_sigmoid(float x) {
  return 1.0f / (1.0f + __expf(-x));
}
__device__ __forceinline__ float fast_tanh(float x) {
  float e = __expf(-2.0f * fabsf(x));
  float t = (1.0f - e) / (1.0f + e);
  return copysignf(t, x);
}
__device__ __forceinline__ v8f wmma_bf16(const AFrag& a, const AFrag& b, v8f c) {
  return __builtin_amdgcn_wmma_f32_16x16x32_bf16(false, a.v, false, b.v,
                                                 (short)0, c, false, false);
}

// ---------------------------------------------------------------------------
// Tensor Data Mover: 1-D contiguous f32 tile, global -> LDS.
// D# per ISA 8.3/8.4: group0 = {count=1 | lds_addr | global_addr | type=2},
// group1 = {data_size=4B, tensor_dim0, tensor_dim1=1, tile_dim0=n}.
// 6-arg builtin (clang-23 / therock-10.0): groups 2/3 + trailing group zeroed.
// ---------------------------------------------------------------------------
__device__ __forceinline__ void tdm_load_f32(const void* lds_dst,
                                             const float* gsrc,
                                             unsigned n) {
  unsigned long long ga = (unsigned long long)(uintptr_t)gsrc;
  unsigned lds = (unsigned)(uintptr_t)lds_dst;   // low 32 bits = LDS offset
  v4u g0;
  g0[0] = 1u;                                          // count=1 (user D#)
  g0[1] = lds;                                         // lds_addr
  g0[2] = (unsigned)(ga & 0xFFFFFFFFu);                // global_addr[31:0]
  g0[3] = (unsigned)((ga >> 32) & 0x01FFFFFFu)         // global_addr[56:32]
        | 0x80000000u;                                 // type=2 ("image")
  unsigned td0 = 0x40000000u;                          // tensor_dim0 (huge, in-bounds)
  v8i_t g1;
  g1[0] = (int)(2u << 16);                             // data_size = 4 bytes
  g1[1] = (int)((td0 & 0xFFFFu) << 16);                // tensor_dim0[15:0]
  g1[2] = (int)((td0 >> 16) | (1u << 16));             // tensor_dim0[31:16] | tensor_dim1=1
  g1[3] = (int)((n & 0xFFFFu) << 16);                  // tile_dim0
  g1[4] = 0;                                           // tile_dim1/2 unused
  g1[5] = (int)td0;                                    // tensor_dim0_stride[31:0]
  g1[6] = 0;
  g1[7] = 0;
  v4i_t z4 = {0, 0, 0, 0};
  v8i_t z8 = {0, 0, 0, 0, 0, 0, 0, 0};
  __builtin_amdgcn_tensor_load_to_lds(g0, g1, z4, z4, z8, 0);
}

// ---------------------------------------------------------------------------
// Kernel 0: pack weight matrices into bf16 WMMA B-fragments.
// 192 blocks x 32 lanes; block = nt*8+s for w_in; s<4 also packs w_rec;
// s==0 && nt<8 packs w_f.
// ---------------------------------------------------------------------------
__global__ __launch_bounds__(32) void efm_pack_kernel(
    const float* __restrict__ w_in,     // [256,384]
    const float* __restrict__ w_rec,    // [128,384]
    const float* __restrict__ w_f,      // [20,128]
    unsigned int* __restrict__ wpHi, unsigned int* __restrict__ wpLo,
    unsigned int* __restrict__ wrHi,
    unsigned int* __restrict__ wfHi, unsigned int* __restrict__ wfLo)
{
  const int lane = threadIdx.x;
  const int blk  = blockIdx.x;
  const int nt = blk >> 3, s = blk & 7;
  const int mrow = lane & 15, half = lane >> 4;
  const int col = nt * 16 + mrow;

  {  // w_in fragment (nt 0..23, s 0..7)

    unsigned int* dh = wpHi + ((size_t)blk * 32 + lane) * 8;
    unsigned int* dl = wpLo + ((size_t)blk * 32 + lane) * 8;
#pragma unroll
    for (int v = 0; v < 8; ++v) {
      int k0 = s * 32 + a_kpair(v, half);
      float f0 = w_in[(size_t)k0 * N3U + col];
      float f1 = w_in[(size_t)(k0 + 1) * N3U + col];
      dh[v] = pack_hi(f0, f1);
      dl[v] = pack_lo(f0, f1);
    }
  }
  if (s < 4) {  // w_rec fragment (nt 0..23, s 0..3), hi only

    unsigned int* d = wrHi + ((size_t)(nt * 4 + s) * 32 + lane) * 8;
#pragma unroll
    for (int v = 0; v < 8; ++v) {
      int k0 = s * 32 + a_kpair(v, half);
      d[v] = pack_hi(w_rec[(size_t)k0 * N3U + col],
                     w_rec[(size_t)(k0 + 1) * N3U + col]);
    }
  }
  if (s == 0 && nt < 8) {  // w_f fragment (nt 0..7), K=20 padded to 32
    const int colu = nt * 16 + mrow;

    unsigned int* dh = wfHi + ((size_t)nt * 32 + lane) * 8;
    unsigned int* dl = wfLo + ((size_t)nt * 32 + lane) * 8;
#pragma unroll
    for (int v = 0; v < 8; ++v) {
      int k0 = a_kpair(v, half);
      float f0 = (k0 < NS) ? w_f[(size_t)k0 * UU + colu] : 0.0f;
      float f1 = (k0 + 1 < NS) ? w_f[(size_t)(k0 + 1) * UU + colu] : 0.0f;
      dh[v] = pack_hi(f0, f1);
      dl[v] = pack_lo(f0, f1);
    }
  }
}

// ---------------------------------------------------------------------------
// Kernel 1: projections. One block per 16-row M-tile of [B*T, *].
// 8 waves; wave s stages A-frags of k-step s into LDS; each wave then owns
// x_proj n-tiles {w, w+8, w+16} and f_seq n-tile {w}.
// ---------------------------------------------------------------------------
__global__ __launch_bounds__(256) void efm_proj_kernel(
    const float* __restrict__ inputs,   // [B*T, 256]
    const float* __restrict__ sigs,     // [B*T, 20]
    const unsigned int* __restrict__ wpHi, const unsigned int* __restrict__ wpLo,
    const unsigned int* __restrict__ wfHi, const unsigned int* __restrict__ wfLo,
    const float* __restrict__ bias,     // [512]
    float* __restrict__ xp,             // [T][B][384]
    float* __restrict__ fs)             // [T][B][128]
{
  __shared__ unsigned int aHiL[8][32][8];   // 8 KB  (A hi frags, k-step major)
  __shared__ unsigned int aLoL[8][32][8];   // 8 KB

  const int tid  = threadIdx.x;
  const int wave = tid >> 5;
  const int lane = tid & 31;
  const int mrow = lane & 15;
  const int half = lane >> 4;

  const int m0 = blockIdx.x * 16;
  const int b  = m0 >> 10;                 // T == 1024
  const int t0 = m0 & (TT - 1);

  // --- stage this block's A-fragments: wave w converts k-step w ------------
  {
    const float* arow = inputs + (size_t)(m0 + mrow) * DD;
#pragma unroll
    for (int v = 0; v < 8; ++v) {
      int k = wave * 32 + a_kpair(v, half);
      float f0 = arow[k], f1 = arow[k + 1];
      aHiL[wave][lane][v] = pack_hi(f0, f1);
      aLoL[wave][lane][v] = pack_lo(f0, f1);
    }
  }
  __syncthreads();

  // --- x_proj: 3 n-tiles per wave, K = 256 ---------------------------------
  DFrag acc[3];
#pragma unroll
  for (int j = 0; j < 3; ++j)
#pragma unroll
    for (int r = 0; r < 8; ++r) acc[j].f[r] = 0.0f;

  for (int s = 0; s < 8; ++s) {
    AFrag aHi, aLo;
#pragma unroll
    for (int v = 0; v < 8; ++v) {
      aHi.u[v] = aHiL[s][lane][v];     // ds_load_b128 x2
      aLo.u[v] = aLoL[s][lane][v];
    }
#pragma unroll
    for (int j = 0; j < 3; ++j) {
      const int nt = wave + 8 * j;
      const unsigned int* ph = wpHi + ((size_t)(nt * 8 + s) * 32 + lane) * 8;
      const unsigned int* pl = wpLo + ((size_t)(nt * 8 + s) * 32 + lane) * 8;
      AFrag wHi, wLo;
#pragma unroll
      for (int v = 0; v < 8; ++v) { wHi.u[v] = ph[v]; wLo.u[v] = pl[v]; }
      // bf16x3: aHi*wHi + aLo*wHi + aHi*wLo  (~fp32 accuracy)
      acc[j].v = wmma_bf16(aHi, wHi, acc[j].v);
      acc[j].v = wmma_bf16(aLo, wHi, acc[j].v);
      acc[j].v = wmma_bf16(aHi, wLo, acc[j].v);
    }
  }
#pragma unroll
  for (int j = 0; j < 3; ++j) {
    const int c0 = (wave + 8 * j) * 16;
#pragma unroll
    for (int r = 0; r < 8; ++r) {
      int t = t0 + r + half * 8;                 // C layout: row = r + 8*half
      xp[((size_t)t * BB + b) * N3U + c0 + mrow] = acc[j].f[r];
    }
  }

  // --- f_seq: 1 n-tile per wave, K = 20 padded to 32 -----------------------
  {
    const float* srow = sigs + (size_t)(m0 + mrow) * NS;
    AFrag aHi, aLo;
#pragma unroll
    for (int v = 0; v < 8; ++v) {
      int k = a_kpair(v, half);
      float f0 = (k < NS) ? srow[k] : 0.0f;
      float f1 = (k + 1 < NS) ? srow[k + 1] : 0.0f;
      aHi.u[v] = pack_hi(f0, f1);
      aLo.u[v] = pack_lo(f0, f1);
    }
    const unsigned int* ph = wfHi + ((size_t)wave * 32 + lane) * 8;
    const unsigned int* pl = wfLo + ((size_t)wave * 32 + lane) * 8;
    AFrag wHi, wLo;
#pragma unroll
    for (int v = 0; v < 8; ++v) { wHi.u[v] = ph[v]; wLo.u[v] = pl[v]; }

    DFrag fa;
#pragma unroll
    for (int r = 0; r < 8; ++r) fa.f[r] = 0.0f;
    fa.v = wmma_bf16(aHi, wHi, fa.v);
    fa.v = wmma_bf16(aLo, wHi, fa.v);
    fa.v = wmma_bf16(aHi, wLo, fa.v);

    const int colu = wave * 16 + mrow;
    const float bfv = bias[UU + colu];           // b_f
#pragma unroll
    for (int r = 0; r < 8; ++r) {
      int t = t0 + r + half * 8;
      fs[((size_t)t * BB + b) * UU + colu] = fast_sigmoid(fa.f[r] + bfv);
    }
  }
}

// ---------------------------------------------------------------------------
// Kernel 2: sequential scan with TDM double buffering.
// Dynamic LDS layout (bytes):
//   gates   16*384 f32   (24576)
//   c       16*128 f32   ( 8192)
//   bias    512    f32   ( 2048)
//   xbuf    2*16*384 f32 (49152)   <- TDM destination
//   fbuf    2*16*128 f32 (16384)   <- TDM destination
//   hHi/hLo 2*16*128 u16 ( 8192)
// ---------------------------------------------------------------------------
#define SCAN_LDS_BYTES ((16 * N3U + 16 * UU + 512 + 2 * 16 * N3U + 2 * 16 * UU) * 4 \
                        + 2 * 16 * UU * 2)

__global__ __launch_bounds__(768) void efm_scan_kernel(
    const unsigned int* __restrict__ wrHi,  // packed W_rec frags
    const float* __restrict__ bias,         // [512]
    const float* __restrict__ xp,           // [T][B][384]
    const float* __restrict__ fs,           // [T][B][128]
    float* __restrict__ out)                // [B][T][128]
{
  extern __shared__ unsigned char smem[];
  float* gatesS = (float*)smem;                         // [16][384]
  float* cS     = gatesS + 16 * N3U;                    // [16][128]
  float* biasS  = cS + 16 * UU;                         // [512]
  float* xbufS  = biasS + 512;                          // [2][16*384]
  float* fbufS  = xbufS + 2 * 16 * N3U;                 // [2][16*128]
  unsigned short* hHiS = (unsigned short*)(fbufS + 2 * 16 * UU);  // [16][128]
  unsigned short* hLoS = hHiS + 16 * UU;

  const int tid  = threadIdx.x;
  const int wave = tid >> 5;            // 0..23
  const int lane = tid & 31;
  const int mrow = lane & 15;
  const int half = lane >> 4;
  const int b0   = blockIdx.x * 16;
  const int c0   = wave * 16;

  // --- W_rec B-fragments into VGPRs for the whole T loop -------------------
  AFrag wfr[4];
#pragma unroll
  for (int s = 0; s < 4; ++s) {
    const unsigned int* p = wrHi + ((size_t)(wave * 4 + s) * 32 + lane) * 8;
#pragma unroll
    for (int v = 0; v < 8; ++v) wfr[s].u[v] = p[v];
  }
  for (int i = tid; i < 4 * UU; i += 768) biasS[i] = bias[i];
  for (int i = tid; i < 16 * UU; i += 768) {
    cS[i] = 0.0f; hHiS[i] = 0; hLoS[i] = 0;
  }
  // preload t=0 tiles via the Tensor Data Mover
  if (wave == 0) {
    tdm_load_f32(xbufS, xp + (size_t)b0 * N3U, 16 * N3U);
    tdm_load_f32(fbufS, fs + (size_t)b0 * UU, 16 * UU);
  }
  __syncthreads();

  for (int t = 0; t < TT; ++t) {
    const int buf = t & 1;
    // ---- gates = (hHi + hLo) @ W_rec  (exact fp32 h, bf16 W) -------------
    DFrag acc;
#pragma unroll
    for (int r = 0; r < 8; ++r) acc.f[r] = 0.0f;
#pragma unroll
    for (int s = 0; s < 4; ++s) {
      AFrag aHi, aLo;
#pragma unroll
      for (int v = 0; v < 8; ++v) {
        int k = s * 32 + a_kpair(v, half);
        aHi.u[v] = *reinterpret_cast<const unsigned int*>(&hHiS[mrow * UU + k]);
        aLo.u[v] = *reinterpret_cast<const unsigned int*>(&hLoS[mrow * UU + k]);
      }
      acc.v = wmma_bf16(aHi, wfr[s], acc.v);
      acc.v = wmma_bf16(aLo, wfr[s], acc.v);
    }
#pragma unroll
    for (int r = 0; r < 8; ++r)
      gatesS[(r + half * 8) * N3U + c0 + mrow] = acc.f[r];

    // ---- TDM: wait for this step's tiles, kick off next step's -----------
    if (wave == 0) {
      __builtin_amdgcn_s_wait_tensorcnt(0);
      if (t + 1 < TT) {
        const int nb = (t + 1) & 1;
        tdm_load_f32(xbufS + nb * 16 * N3U,
                     xp + ((size_t)(t + 1) * BB + b0) * N3U, 16 * N3U);
        tdm_load_f32(fbufS + nb * 16 * UU,
                     fs + ((size_t)(t + 1) * BB + b0) * UU, 16 * UU);
      }
    }
    __syncthreads();

    // ---- fused gate nonlinearity + state update --------------------------
    const float* xb = xbufS + buf * 16 * N3U;
    const float* fb = fbufS + buf * 16 * UU;
    for (int e = tid; e < 16 * UU; e += 768) {
      const int r = e >> 7;
      const int u = e & (UU - 1);
      const float i_g  = fast_sigmoid(gatesS[r * N3U + u] + xb[r * N3U + u] + biasS[u]);
      const float chat = fast_tanh(gatesS[r * N3U + UU + u] + xb[r * N3U + UU + u]
                                   + biasS[2 * UU + u]);
      const float o_g  = fast_sigmoid(gatesS[r * N3U + 2 * UU + u]
                                      + xb[r * N3U + 2 * UU + u] + biasS[3 * UU + u]);
      const float c_new = fb[r * UU + u] * cS[r * UU + u] + i_g * chat;
      cS[r * UU + u] = c_new;
      const float h_new = o_g * fast_tanh(c_new);

      out[(((size_t)(b0 + r)) * TT + t) * UU + u] = h_new;

      const unsigned short hh = bf16_rn(h_new);
      hHiS[r * UU + u] = hh;
      hLoS[r * UU + u] = bf16_rn(h_new - bf16_f32(hh));
    }
    __syncthreads();
  }
}

extern "C" void kernel_launch(void* const* d_in, const int* in_sizes, int n_in,
                              void* d_out, int out_size, void* d_ws, size_t ws_size,
                              hipStream_t stream) {
  const float* inputs = (const float*)d_in[0];   // [B,T,D]
  const float* sigs   = (const float*)d_in[1];   // [B,T,SIG]
  const float* w_in   = (const float*)d_in[2];   // [D,3U]
  const float* w_rec  = (const float*)d_in[3];   // [U,3U]
  const float* w_f    = (const float*)d_in[4];   // [SIG,U]
  const float* bias   = (const float*)d_in[5];   // [4U]
  float* out = (float*)d_out;

  float* xp = (float*)d_ws;                              // [T][B][384] f32
  float* fs = xp + (size_t)BB * TT * N3U;                // [T][B][128] f32
  unsigned int* wpHi = (unsigned int*)(fs + (size_t)BB * TT * UU);
  unsigned int* wpLo = wpHi + WPIN_WORDS;
  unsigned int* wrHi = wpLo + WPIN_WORDS;
  unsigned int* wfHi = wrHi + WREC_WORDS;
  unsigned int* wfLo = wfHi + WF_WORDS;
  // ws needed: 512 MiB (xp+fs) + ~508 KiB packed weights

  efm_pack_kernel<<<24 * 8, 32, 0, stream>>>(w_in, w_rec, w_f,
                                             wpHi, wpLo, wrHi, wfHi, wfLo);
  efm_proj_kernel<<<(BB * TT) / 16, 256, 0, stream>>>(inputs, sigs,
                                                      wpHi, wpLo, wfHi, wfLo,
                                                      bias, xp, fs);
  efm_scan_kernel<<<BB / 16, 768, SCAN_LDS_BYTES, stream>>>(wrHi, bias, xp, fs,
                                                            out);
}
